// GRNN_5720896438852
// MI455X (gfx1250) — matrix operand order repaired and checked by reading, not compiled, not run on hardware
//
#include <hip/hip_runtime.h>
#include <hip/hip_bf16.h>
#include <stdint.h>

// ---------------------------------------------------------------------------
// CDNA5 / gfx1250 implementation:
//   GC1:  Xn = bf16(features * rsqrt(deg_out));  proj = Xn @ W1  (v_wmma bf16)
//         agg[dst] += proj[src]  (L2-resident float atomics)
//         x1 = relu(agg * rsqrt(deg_in) + b1); re-scale+bf16 for layer 2
//   GC2:  same with W2
//   LSTM: Xproj = x2 @ W_ih^T hoisted as one WMMA GEMM (no serial dep),
//         then a single persistent 128-thread block runs the 20000-step
//         recurrence with W_hh in registers and h/gates in LDS.
//   FC:   out = hs @ W_fc + b_fc
// ---------------------------------------------------------------------------

typedef __attribute__((ext_vector_type(16))) __bf16 v16bf;
typedef __attribute__((ext_vector_type(8)))  float  v8f;
typedef unsigned short ushort_t;

#define N_NODES 20000
#define N_EDGES 640000
#define IN_F    128
#define H1F     64
#define H2F     64
#define LSTM_H  32
#define N_CLS   10

__device__ __forceinline__ ushort_t f2bf(float x) {
    unsigned int u = __float_as_uint(x);
    u += 0x7FFFu + ((u >> 16) & 1u);   // round-to-nearest-even
    return (ushort_t)(u >> 16);
}

// ---------------- utility kernels ----------------
__global__ void fill_f32(float* __restrict__ p, float v, int n) {
    int i = blockIdx.x * blockDim.x + threadIdx.x;
    if (i < n) p[i] = v;
}

__global__ void deg_count(const int* __restrict__ src, const int* __restrict__ dst,
                          float* __restrict__ dego, float* __restrict__ degi, int nE) {
    int e = blockIdx.x * blockDim.x + threadIdx.x;
    if (e < nE) {
        atomicAdd(&dego[src[e]], 1.0f);
        atomicAdd(&degi[dst[e]], 1.0f);
    }
}

__global__ void rsq_finalize(float* __restrict__ p, int n) {
    int i = blockIdx.x * blockDim.x + threadIdx.x;
    if (i < n) p[i] = rsqrtf(fmaxf(p[i], 1.0f));
}

// Pack weights to bf16.  B operands are stored transposed as [N,K] so every
// WMMA lane load is contiguous 128-bit.
__global__ void pack_weights(const float* __restrict__ W1, const float* __restrict__ W2,
                             const float* __restrict__ Wih,
                             ushort_t* __restrict__ W1t, ushort_t* __restrict__ W2t,
                             ushort_t* __restrict__ WihB) {
    int i = blockIdx.x * blockDim.x + threadIdx.x;
    if (i < IN_F * H1F) {                       // W1: [128,64] -> W1t [64,128]
        int k = i >> 6, n = i & 63;
        W1t[n * IN_F + k] = f2bf(W1[i]);
    }
    if (i < H1F * H2F) {                        // W2: [64,64] -> W2t [64,64]
        int k = i >> 6, n = i & 63;
        W2t[n * H1F + k] = f2bf(W2[i]);
    }
    if (i < 4 * LSTM_H * H2F) {                 // W_ih: [128,64] is already [N,K]
        WihB[i] = f2bf(Wih[i]);
    }
}

__global__ void feat_prep(const float* __restrict__ x, const float* __restrict__ rsqo,
                          ushort_t* __restrict__ out, int n) {
    int i = blockIdx.x * blockDim.x + threadIdx.x;
    if (i < n) out[i] = f2bf(x[i] * rsqo[i >> 7]);   // F = 128
}

// ---------------- WMMA bf16 GEMM: C[M,N] = A[M,K] @ B[K,N] ----------------
// A row-major bf16 (lda=K); Bt is B transposed, [N,K] bf16; C row-major f32.
// One wave per 16x16 tile; K loop in steps of 32; all operand loads are b128.
__global__ void __launch_bounds__(32)
wmma_gemm_bf16(const ushort_t* __restrict__ A, const ushort_t* __restrict__ Bt,
               float* __restrict__ C, int M, int N, int K) {
    int lane = threadIdx.x & 31;
    int n0 = blockIdx.x * 16;
    int m0 = blockIdx.y * 16;
    int row = lane & 15;
    int hi  = lane >> 4;

    v8f acc = {0.f, 0.f, 0.f, 0.f, 0.f, 0.f, 0.f, 0.f};
    const ushort_t* arow = A  + (size_t)(m0 + row) * K;
    const ushort_t* brow = Bt + (size_t)(n0 + row) * K;

    for (int kk = 0; kk < K; kk += 32) {
        union { uint4 q[2]; v16bf v; } ua, ub;
        // A frag: lanes 0-15 hold K=kk..kk+7 & kk+16..kk+23; lanes 16-31 the other halves
        ua.q[0] = *reinterpret_cast<const uint4*>(arow + kk + hi * 8);
        ua.q[1] = *reinterpret_cast<const uint4*>(arow + kk + hi * 8 + 16);
        // B frag: lanes 0-15 hold K=kk..kk+7 & kk+16..kk+23 of column n; lanes 16-31 the other halves
        ub.q[0] = *reinterpret_cast<const uint4*>(brow + kk + hi * 8);
        ub.q[1] = *reinterpret_cast<const uint4*>(brow + kk + hi * 8 + 16);
        acc = __builtin_amdgcn_wmma_f32_16x16x32_bf16(false, ua.v, false, ub.v,
                                                      (short)0, acc, false, false);
    }
    // C/D layout: VGPR r -> row m0 + hi*8 + r, column n0 + (lane&15)
    float* crow = C + (size_t)(m0 + hi * 8) * N + n0 + row;
#pragma unroll
    for (int r = 0; r < 8; ++r) crow[(size_t)r * N] = acc[r];
}

// ---------------- edge scatter-add (L2 resident) ----------------
__global__ void edge_agg(const int* __restrict__ src, const int* __restrict__ dst,
                         const float* __restrict__ h, float* __restrict__ agg, int nE) {
    long long i = (long long)blockIdx.x * blockDim.x + threadIdx.x;
    int e = (int)(i >> 6);          // F = 64
    int f = (int)(i & 63);
    if (e < nE) {
        __builtin_prefetch(&src[e + 4096], 0, 1);   // gfx1250 global_prefetch_b8
        __builtin_prefetch(&dst[e + 4096], 0, 1);
        atomicAdd(&agg[(size_t)dst[e] * 64 + f], h[(size_t)src[e] * 64 + f]);
    }
}

// x = relu(agg * rsqrt(deg_in) + b); optionally rescale by rsqrt(deg_out)
// for the next GraphConv; emit bf16 ready for the next WMMA GEMM.
__global__ void gc_finalize(const float* __restrict__ agg, const float* __restrict__ rsqi,
                            const float* __restrict__ bias, const float* __restrict__ rsqo,
                            ushort_t* __restrict__ outb, int n) {
    int i = blockIdx.x * blockDim.x + threadIdx.x;
    if (i >= n) return;
    int node = i >> 6;              // F = 64
    int f = i & 63;
    float v = fmaxf(agg[i] * rsqi[node] + bias[f], 0.0f);
    if (rsqo) v *= rsqo[node];
    outb[i] = f2bf(v);
}

// ---------------- persistent single-block LSTM recurrence ----------------
// xproj[t,g] = x_t @ W_ih^T already computed by WMMA.  Thread g owns gate g;
// W_hh row in registers; h and activated gates in LDS; threads 0..31 own c[j].
__global__ void __launch_bounds__(128)
lstm_kernel(const float* __restrict__ xproj, const float* __restrict__ Whh,
            const float* __restrict__ b_ih, const float* __restrict__ b_hh,
            float* __restrict__ hs, int T) {
    __shared__ float sh[LSTM_H];
    __shared__ float sg[4 * LSTM_H];
    int g = threadIdx.x;            // 0..127
    float wr[LSTM_H];
#pragma unroll
    for (int k = 0; k < LSTM_H; ++k) wr[k] = Whh[g * LSTM_H + k];
    float bsum = b_ih[g] + b_hh[g];
    float c = 0.0f;
    if (g < LSTM_H) sh[g] = 0.0f;
    __syncthreads();

    for (int t = 0; t < T; ++t) {
        float acc = xproj[(size_t)t * 128 + g] + bsum;
#pragma unroll
        for (int k = 0; k < LSTM_H; ++k) acc += wr[k] * sh[k];
        // gates order i,f,g,o : tanh for gate block 2, sigmoid otherwise
        float a = ((g >> 5) == 2) ? tanhf(acc) : (1.0f / (1.0f + __expf(-acc)));
        sg[g] = a;
        __syncthreads();
        if (g < LSTM_H) {
            c = sg[LSTM_H + g] * c + sg[g] * sg[2 * LSTM_H + g];
            float h = sg[3 * LSTM_H + g] * tanhf(c);
            sh[g] = h;
            hs[(size_t)t * LSTM_H + g] = h;
        }
        __syncthreads();
    }
}

// ---------------- final FC ----------------
__global__ void fc_kernel(const float* __restrict__ hs, const float* __restrict__ Wfc,
                          const float* __restrict__ bfc, float* __restrict__ out, int n) {
    __shared__ float sW[LSTM_H * N_CLS];
    __shared__ float sb[N_CLS];
    int t = threadIdx.x;
    for (int i = t; i < LSTM_H * N_CLS; i += blockDim.x) sW[i] = Wfc[i];
    if (t < N_CLS) sb[t] = bfc[t];
    __syncthreads();
    int idx = blockIdx.x * blockDim.x + t;
    if (idx >= n) return;
    int node = idx / N_CLS;
    int cls  = idx - node * N_CLS;
    const float* hrow = hs + (size_t)node * LSTM_H;
    float acc = sb[cls];
#pragma unroll
    for (int k = 0; k < LSTM_H; ++k) acc += hrow[k] * sW[k * N_CLS + cls];
    out[idx] = acc;
}

// ---------------------------------------------------------------------------
extern "C" void kernel_launch(void* const* d_in, const int* in_sizes, int n_in,
                              void* d_out, int out_size, void* d_ws, size_t ws_size,
                              hipStream_t stream) {
    (void)in_sizes; (void)n_in; (void)out_size; (void)ws_size;
    const float* features = (const float*)d_in[0];
    const int*   src      = (const int*)  d_in[1];
    const int*   dst      = (const int*)  d_in[2];
    const float* W1       = (const float*)d_in[3];
    const float* b1       = (const float*)d_in[4];
    const float* W2       = (const float*)d_in[5];
    const float* b2       = (const float*)d_in[6];
    const float* Wih      = (const float*)d_in[7];
    const float* Whh      = (const float*)d_in[8];
    const float* b_ih     = (const float*)d_in[9];
    const float* b_hh     = (const float*)d_in[10];
    const float* Wfc      = (const float*)d_in[11];
    const float* bfc      = (const float*)d_in[12];
    float* out = (float*)d_out;

    // workspace layout (256B aligned)
    char* ws = (char*)d_ws;
    size_t off = 0;
    auto take = [&](size_t bytes) { size_t o = off; off = (off + bytes + 255) & ~(size_t)255; return o; };
    float*    dego  = (float*)   (ws + take((size_t)2 * N_NODES * 4)); // [dego | degi]
    float*    degi  = dego + N_NODES;
    ushort_t* Abf   = (ushort_t*)(ws + take((size_t)N_NODES * IN_F * 2));
    ushort_t* W1t   = (ushort_t*)(ws + take((size_t)IN_F * H1F * 2));
    ushort_t* W2t   = (ushort_t*)(ws + take((size_t)H1F * H2F * 2));
    ushort_t* WihB  = (ushort_t*)(ws + take((size_t)4 * LSTM_H * H2F * 2));
    float*    proj  = (float*)   (ws + take((size_t)N_NODES * H1F * 4));
    float*    agg   = (float*)   (ws + take((size_t)N_NODES * H1F * 4));
    float*    xproj = (float*)   (ws + take((size_t)N_NODES * 4 * LSTM_H * 4));
    float*    hs    = (float*)   (ws + take((size_t)N_NODES * LSTM_H * 4));

    const int B = 256;
    auto blocks = [](long long n, int b) { return (unsigned)((n + b - 1) / b); };

    // degrees -> rsqrt
    fill_f32<<<blocks(2 * N_NODES, B), B, 0, stream>>>(dego, 0.0f, 2 * N_NODES);
    deg_count<<<blocks(N_EDGES, B), B, 0, stream>>>(src, dst, dego, degi, N_EDGES);
    rsq_finalize<<<blocks(2 * N_NODES, B), B, 0, stream>>>(dego, 2 * N_NODES);

    // weight packing + scaled-feature bf16
    pack_weights<<<blocks(IN_F * H1F, B), B, 0, stream>>>(W1, W2, Wih, W1t, W2t, WihB);
    feat_prep<<<blocks((long long)N_NODES * IN_F, B), B, 0, stream>>>(
        features, dego, Abf, N_NODES * IN_F);

    // GraphConv 1
    wmma_gemm_bf16<<<dim3(H1F / 16, N_NODES / 16), 32, 0, stream>>>(
        Abf, W1t, proj, N_NODES, H1F, IN_F);
    fill_f32<<<blocks((long long)N_NODES * H1F, B), B, 0, stream>>>(agg, 0.0f, N_NODES * H1F);
    edge_agg<<<blocks((long long)N_EDGES * 64, B), B, 0, stream>>>(src, dst, proj, agg, N_EDGES);
    gc_finalize<<<blocks((long long)N_NODES * H1F, B), B, 0, stream>>>(
        agg, degi, b1, dego, Abf, N_NODES * H1F);

    // GraphConv 2
    wmma_gemm_bf16<<<dim3(H2F / 16, N_NODES / 16), 32, 0, stream>>>(
        Abf, W2t, proj, N_NODES, H2F, H1F);
    fill_f32<<<blocks((long long)N_NODES * H2F, B), B, 0, stream>>>(agg, 0.0f, N_NODES * H2F);
    edge_agg<<<blocks((long long)N_EDGES * 64, B), B, 0, stream>>>(src, dst, proj, agg, N_EDGES);
    gc_finalize<<<blocks((long long)N_NODES * H2F, B), B, 0, stream>>>(
        agg, degi, b2, nullptr, Abf, N_NODES * H2F);

    // hoisted LSTM input projection: [20000,64] @ [64,128]
    wmma_gemm_bf16<<<dim3((4 * LSTM_H) / 16, N_NODES / 16), 32, 0, stream>>>(
        Abf, WihB, xproj, N_NODES, 4 * LSTM_H, H2F);

    // serial recurrence + FC
    lstm_kernel<<<1, 128, 0, stream>>>(xproj, Whh, b_ih, b_hh, hs, N_NODES);
    fc_kernel<<<blocks((long long)N_NODES * N_CLS, B), B, 0, stream>>>(
        hs, Wfc, bfc, out, N_NODES * N_CLS);
}
